// CWDiscriminator_80882824118587
// MI455X (gfx1250) — matrix operand order repaired and checked by reading, not compiled, not run on hardware
//
#include <hip/hip_runtime.h>
#include <stdint.h>

// ---------------------------------------------------------------------------
// CWDiscriminator fused kernel (v4): per-class 2-layer MLP + linear head.
//   h1 = relu(x @ W1[c] + b1[c]);  h2 = relu(h1 @ W2[c] + b2[c])
//   out[b,c] = h2 . W3[c] + b3[c]
// bf16 WMMA, fp32 accumulate. One 16-row M-tile per wave; two N-tiles per
// step share the register-resident A-fragments (two independent WMMA chains).
// Weights live in LDS pre-swizzled so every B fragment is 2x ds_load_b128
// (bank-conflict-free, immediate offsets). h1 staging uses a 528-byte padded
// row stride so layer-2 A-fragment b128 loads are conflict-free. Layer 3 is
// folded into the layer-2 epilogue via a half-wave shuffle reduction.
// ---------------------------------------------------------------------------

typedef __attribute__((ext_vector_type(16))) __bf16 v16bf;
typedef __attribute__((ext_vector_type(8)))  float  v8f;

union Frag { v16bf v; uint4 q[2]; uint32_t u[8]; };

#define FDIM 256
#define NCLS 19
#define BATCH 16384
#define WG_PER_CLASS 32
#define ROWS_PER_WG (BATCH / WG_PER_CLASS)   // 512
#define NWAVES 4
#define NTHREADS (NWAVES * 32)
#define MT_PER_WG (ROWS_PER_WG / 16)         // 32 m-tiles, 8 per wave

#define HROW 264                             // padded h-staging row stride (u16)
#define HTILE (16 * HROW)                    // 4224 u16 = 8448 B per wave

// LDS layout (dynamic shared, 299,008 B < 320 KB/WGP)
#define OFF_W1 0         // 131072 B : W1[c] bf16, b128-fragment swizzled
#define OFF_W2 131072    // 131072 B : W2[c] bf16, b128-fragment swizzled
#define OFF_H  262144    //  33792 B : per-wave 16 x 264 u16 h1 staging (padded)
#define OFF_B1 295936    //   1024 B
#define OFF_B2 296960    //   1024 B
#define OFF_W3 297984    //   1024 B
#define SMEM_BYTES 299008

__device__ __forceinline__ uint32_t f2bf(float f) {       // fp32 -> bf16 (RNE)
  uint32_t x = __builtin_bit_cast(uint32_t, f);
  return (x + 0x7FFFu + ((x >> 16) & 1u)) >> 16;
}

__global__ void cw_disc_fused_kernel(const float* __restrict__ inputs,
                                     const float* __restrict__ W1,
                                     const float* __restrict__ b1,
                                     const float* __restrict__ W2,
                                     const float* __restrict__ b2,
                                     const float* __restrict__ W3,
                                     const float* __restrict__ b3,
                                     float* __restrict__ out) {
  extern __shared__ char smem[];
  uint32_t* w1buf = (uint32_t*)(smem + OFF_W1);
  uint32_t* w2buf = (uint32_t*)(smem + OFF_W2);
  uint16_t* hbase = (uint16_t*)(smem + OFF_H);
  float*    b1s   = (float*)(smem + OFF_B1);
  float*    b2s   = (float*)(smem + OFF_B2);
  float*    w3s   = (float*)(smem + OFF_W3);

  const int tid  = threadIdx.x;
  const int lane = tid & 31;
  const int wave = tid >> 5;
  const int wg   = blockIdx.x;   // batch slab
  const int c    = blockIdx.y;   // class

  const float* W1c = W1 + (size_t)c * FDIM * FDIM;
  const float* W2c = W2 + (size_t)c * FDIM * FDIM;

  // ---- Stage W1[c], W2[c] into LDS, swizzled for b128 fragment loads ----
  // Source iteration i = nt*2048 + kk*256 + v*32 + l (l fastest: coalesced
  // global reads of consecutive n). Destination dword:
  //   dst = (nt*8+kk)*256 + (v>>2)*128 + l*4 + (v&3)
  // so lane l's fragment = dwords [base+4l .. +3] and [base+128+4l .. +3]:
  // two 16B chunks, each half-wave b128 covering all 64 banks once.
  // Element mapping (B 32x16 bf16): n = nt*16+(l&15);
//   k = kk*32 + ((v&4)?16:0) + ((l&16)?8:0) + 2(v&3); word packs (K=k, K=k+1).
  for (int i = tid; i < 32768; i += NTHREADS) {
    int l  = i & 31;
    int v  = (i >> 5) & 7;
    int kk = (i >> 8) & 7;
    int nt = i >> 11;
    int n  = nt * 16 + (l & 15);
    int k  = kk * 32 + ((v & 4) ? 16 : 0) + ((l & 16) ? 8 : 0) + 2 * (v & 3);
    int dst = (nt * 8 + kk) * 256 + (v >> 2) * 128 + l * 4 + (v & 3);
    w1buf[dst] = f2bf(W1c[k * FDIM + n]) | (f2bf(W1c[(k + 1) * FDIM + n]) << 16);
    w2buf[dst] = f2bf(W2c[k * FDIM + n]) | (f2bf(W2c[(k + 1) * FDIM + n]) << 16);
  }
  for (int i = tid; i < FDIM; i += NTHREADS) {
    b1s[i] = b1[c * FDIM + i];
    b2s[i] = b2[c * FDIM + i];
    w3s[i] = W3[c * FDIM + i];
  }
  __syncthreads();

  // A 16x32 bf16 layout: lane&15 = row m; lanes 16-31 K-offset +8;
  // VGPR v: K = (v>=4?16:0) + kg + 2*(v&3), packed pairs.
  const int row  = lane & 15;
  const int hi   = (lane & 16) ? 1 : 0;
  const int kgA  = hi * 8;
  const int mhi  = hi * 8;     // C/D layout: lanes 16-31 hold rows M=8..15
  const int ncl  = lane & 15;  // C/D layout: column = lane&15
  uint16_t* hst  = hbase + wave * HTILE;
  const float b3c = b3[c];
  const uint4* w1q = (const uint4*)w1buf;
  const uint4* w2q = (const uint4*)w2buf;

#pragma unroll 1
  for (int mt = wave; mt < MT_PER_WG; mt += NWAVES) {
    const int base_b = wg * ROWS_PER_WG + mt * 16;

    // ---- Gather layer-1 A fragments from inputs (B,F,C), fp32->bf16 ----
    Frag A[8];
    {
      const float* xp = inputs + (size_t)(base_b + row) * (FDIM * NCLS) + c;
#pragma unroll
      for (int kk = 0; kk < 8; ++kk) {
#pragma unroll
        for (int v = 0; v < 8; ++v) {
          int k = kk * 32 + ((v & 4) ? 16 : 0) + kgA + 2 * (v & 3);
          A[kk].u[v] = f2bf(xp[(size_t)k * NCLS]) |
                       (f2bf(xp[(size_t)(k + 1) * NCLS]) << 16);
        }
      }
    }

    // ---- Layer 1: two N-tiles per step share A; relu -> LDS bf16 ----
#pragma unroll 1
    for (int nt = 0; nt < 16; nt += 2) {
      v8f acc0 = {0.f, 0.f, 0.f, 0.f, 0.f, 0.f, 0.f, 0.f};
      v8f acc1 = {0.f, 0.f, 0.f, 0.f, 0.f, 0.f, 0.f, 0.f};
      const uint4* q0 = w1q + nt * 512 + lane;        // (nt*8)*64 uint4
      const uint4* q1 = q0 + 512;                     // next N-tile
      int ncol0 = nt * 16 + ncl;
      int ncol1 = ncol0 + 16;
      float bias0 = b1s[ncol0];
      float bias1 = b1s[ncol1];
#pragma unroll
      for (int kk = 0; kk < 8; ++kk) {
        Frag Bf0, Bf1;
        Bf0.q[0] = q0[kk * 64];
        Bf0.q[1] = q0[kk * 64 + 32];
        Bf1.q[0] = q1[kk * 64];
        Bf1.q[1] = q1[kk * 64 + 32];
        acc0 = __builtin_amdgcn_wmma_f32_16x16x32_bf16(
            false, A[kk].v, false, Bf0.v, (short)0, acc0, false, false);
        acc1 = __builtin_amdgcn_wmma_f32_16x16x32_bf16(
            false, A[kk].v, false, Bf1.v, (short)0, acc1, false, false);
      }
#pragma unroll
      for (int j = 0; j < 8; ++j) {
        float h0 = fmaxf(acc0[j] + bias0, 0.f);
        float h1 = fmaxf(acc1[j] + bias1, 0.f);
        hst[(mhi + j) * HROW + ncol0] = (uint16_t)f2bf(h0);
        hst[(mhi + j) * HROW + ncol1] = (uint16_t)f2bf(h1);
      }
    }
    asm volatile("s_wait_dscnt 0" ::: "memory");  // h1 visible wave-wide

    // ---- Layer-2 A fragments: two b128 loads per chunk from padded LDS ----
    Frag A2[8];
    {
      const char* hrb = (const char*)(hst + row * HROW) + kgA * 2;
#pragma unroll
      for (int kk = 0; kk < 8; ++kk) {
        A2[kk].q[0] = *(const uint4*)(hrb + kk * 64);
        A2[kk].q[1] = *(const uint4*)(hrb + kk * 64 + 32);
      }
    }

    // ---- Layer 2 with layer-3 folded: r[j] += relu(h2)*w3[ncol] ----
    float r[8] = {0.f, 0.f, 0.f, 0.f, 0.f, 0.f, 0.f, 0.f};
#pragma unroll 1
    for (int nt = 0; nt < 16; nt += 2) {
      v8f acc0 = {0.f, 0.f, 0.f, 0.f, 0.f, 0.f, 0.f, 0.f};
      v8f acc1 = {0.f, 0.f, 0.f, 0.f, 0.f, 0.f, 0.f, 0.f};
      const uint4* q0 = w2q + nt * 512 + lane;
      const uint4* q1 = q0 + 512;
      int ncol0 = nt * 16 + ncl;
      int ncol1 = ncol0 + 16;
      float bias0 = b2s[ncol0];
      float bias1 = b2s[ncol1];
      float w30 = w3s[ncol0];
      float w31 = w3s[ncol1];
#pragma unroll
      for (int kk = 0; kk < 8; ++kk) {
        Frag Bf0, Bf1;
        Bf0.q[0] = q0[kk * 64];
        Bf0.q[1] = q0[kk * 64 + 32];
        Bf1.q[0] = q1[kk * 64];
        Bf1.q[1] = q1[kk * 64 + 32];
        acc0 = __builtin_amdgcn_wmma_f32_16x16x32_bf16(
            false, A2[kk].v, false, Bf0.v, (short)0, acc0, false, false);
        acc1 = __builtin_amdgcn_wmma_f32_16x16x32_bf16(
            false, A2[kk].v, false, Bf1.v, (short)0, acc1, false, false);
      }
#pragma unroll
      for (int j = 0; j < 8; ++j) {
        r[j] += fmaxf(acc0[j] + bias0, 0.f) * w30 +
                fmaxf(acc1[j] + bias1, 0.f) * w31;
      }
    }

    // ---- Reduce r[j] across each 16-lane half; write out[b,c] ----
    // Lane half lo holds rows 0..7 (j), half hi rows 8..15 (8+j).
#pragma unroll
    for (int j = 0; j < 8; ++j) {
      float v0 = r[j];
      v0 += __shfl_xor(v0, 1, 32);
      v0 += __shfl_xor(v0, 2, 32);
      v0 += __shfl_xor(v0, 4, 32);
      v0 += __shfl_xor(v0, 8, 32);
      if (ncl == j) {
        out[(size_t)(base_b + mhi + j) * NCLS + c] = v0 + b3c;
      }
    }
  }
}

extern "C" void kernel_launch(void* const* d_in, const int* in_sizes, int n_in,
                              void* d_out, int out_size, void* d_ws, size_t ws_size,
                              hipStream_t stream) {
  const float* inputs = (const float*)d_in[0];
  const float* W1 = (const float*)d_in[1];
  const float* b1 = (const float*)d_in[2];
  const float* W2 = (const float*)d_in[3];
  const float* b2 = (const float*)d_in[4];
  const float* W3 = (const float*)d_in[5];
  const float* b3 = (const float*)d_in[6];
  float* out = (float*)d_out;

  dim3 grid(WG_PER_CLASS, NCLS);   // 32 batch slabs x 19 classes
  dim3 block(NTHREADS);            // 4 waves (wave32)
  cw_disc_fused_kernel<<<grid, block, SMEM_BYTES, stream>>>(
      inputs, W1, b1, W2, b2, W3, b3, out);
}
